// NeuroSAT_34849364640214
// MI455X (gfx1250) — matrix operand order repaired and checked by reading, not compiled, not run on hardware
//
#include <hip/hip_runtime.h>
#include <hip/hip_bf16.h>

// ---------------------------------------------------------------------------
// NeuroSAT message passing on MI455X (gfx1250, wave32, WMMA bf16)
// ---------------------------------------------------------------------------

#define FM 256
#define ROUNDS 8
#define NV 10000
#define NL 20000
#define NC 42000
#define NE 126000

typedef __attribute__((ext_vector_type(16))) __bf16 v16bf;
typedef __attribute__((ext_vector_type(4)))  __bf16 v4bf;
typedef __attribute__((ext_vector_type(8)))  float  v8f;

__device__ __forceinline__ float sigf(float x) {
    return 1.0f / (1.0f + __expf(-x));
}

// ---- elementwise kernels ---------------------------------------------------

// convert f32 [K,N] row-major -> bf16 [N,K] (transposed, WMMA-B friendly)
__global__ void k_convT(const float* __restrict__ src, __bf16* __restrict__ dst,
                        int K, int N) {
    long i = (long)blockIdx.x * blockDim.x + threadIdx.x;
    if (i >= (long)K * N) return;
    int k = (int)(i / N);
    int n = (int)(i % N);
    dst[(size_t)n * K + k] = (__bf16)src[i];
}

__global__ void k_zero_f32(float* __restrict__ p, long n) {
    long i = (long)blockIdx.x * blockDim.x + threadIdx.x;
    if (i < n) p[i] = 0.0f;
}

// dst[r, :] = src[:] * scale for r in [0, rows)
__global__ void k_bcast_row(const float* __restrict__ src, float scale,
                            float* __restrict__ dst, int rows) {
    long i = (long)blockIdx.x * blockDim.x + threadIdx.x;
    if (i < (long)rows * FM) dst[i] = src[i % FM] * scale;
}

// segment-sum scatter: dst[dstIdx[e], :] += src[srcIdx[e], :]
__global__ void k_scatter_add(const float* __restrict__ src,
                              const int* __restrict__ srcIdx,
                              const int* __restrict__ dstIdx,
                              float* __restrict__ dst, int nEdges) {
    int e = blockIdx.x;
    if (e >= nEdges) return;
    const float* s = src + (size_t)srcIdx[e] * FM;
    float*       d = dst + (size_t)dstIdx[e] * FM;
    for (int c = threadIdx.x; c < FM; c += blockDim.x)
        atomicAdd(&d[c], s[c]);
}

// Keras LSTMCell: z = [i | f | g | o] columns of width FM each in Z[M, 4*FM]
__global__ void k_lstm(const float* __restrict__ Z, float* __restrict__ h,
                       float* __restrict__ c, int M) {
    long idx = (long)blockIdx.x * blockDim.x + threadIdx.x;
    if (idx >= (long)M * FM) return;
    int m = (int)(idx / FM);
    int j = (int)(idx % FM);
    const float* z = Z + (size_t)m * (4 * FM);
    float zi = z[j], zf = z[FM + j], zg = z[2 * FM + j], zo = z[3 * FM + j];
    float cc = sigf(zf) * c[idx] + sigf(zi) * tanhf(zg);
    c[idx] = cc;
    h[idx] = sigf(zo) * tanhf(cc);
}

// out[m] = sum_k x[m,k] * w[k] + b[0]   (one wave32 per row)
__global__ void k_vote_out(const float* __restrict__ x, const float* __restrict__ w,
                           const float* __restrict__ b, float* __restrict__ out,
                           int M, int K) {
    int gt   = blockIdx.x * blockDim.x + threadIdx.x;
    int wave = gt >> 5;
    int lane = threadIdx.x & 31;
    if (wave >= M) return;
    const float* xr = x + (size_t)wave * K;
    float s = 0.0f;
    for (int k = lane; k < K; k += 32) s += xr[k] * w[k];
    for (int o = 16; o > 0; o >>= 1) s += __shfl_down(s, o, 32);
    if (lane == 0) out[wave] = s + b[0];
}

// ---- multi-pair WMMA GEMM --------------------------------------------------
// C[M,N] = sum_p A_p[M,K_p](f32->bf16 on stage) @ B_p (bf16, PRE-TRANSPOSED
// [N][K_p]) + bias[N], optional ReLU, per-pair rotated-row read of A
// (row=(m+rot) mod rowMod) implements the "flipped" literal permutation.
// Fusing up to 3 pairs keeps LSTM gate pre-activations in WMMA accumulators.
//
// Tile: 128x128 per 256-thread block; 8 waves in 4(M) x 2(N); each wave owns
// 32x64 = 2x4 v_wmma_f32_16x16x32_bf16 fragments; BK = 32. DOUBLE-BUFFERED
// LDS ping-pong: one barrier per k-step. A's next-tile loads are issued
// before the WMMA block (HBM latency hidden by compute); B's next-tile loads
// are issued after it (L2-resident weights, short liveness, no spills).

#define BM 128
#define BN 128
#define BK 32
#define LDP (BK + 8)   // padded LDS K-stride (elements)

struct GemmArgs {
    const float*  A[3];
    const __bf16* B[3];      // transposed: [N][K_p]
    int K[3];
    int rot[3];
    int rowMod[3];
    int M, N;
    const float* bias;
    float* C;
    int doRelu;
};

template <bool GUARD>
__device__ __forceinline__ void gemm_pairs(const GemmArgs& g,
                                           __bf16 (&As)[2][BM][LDP],
                                           __bf16 (&Bt)[2][BN][LDP]) {
    const int tid   = threadIdx.x;
    const int wave  = tid >> 5;
    const int lane  = tid & 31;
    const int waveM = wave & 3;           // 4 wave-rows   (32 rows each)
    const int waveN = wave >> 2;          // 2 wave-cols   (64 cols each)
    const int half  = lane >> 4;
    const int l16   = lane & 15;

    const int blockM = blockIdx.y * BM;
    const int blockN = blockIdx.x * BN;

    v8f acc[2][4];
#pragma unroll
    for (int mi = 0; mi < 2; mi++)
#pragma unroll
        for (int ni = 0; ni < 4; ni++)
#pragma unroll
            for (int r = 0; r < 8; r++) acc[mi][ni][r] = 0.0f;

    union Frag { uint4 q[2]; v16bf v; };

    const int aC4 = (tid & 7) * 4;        // A stage: K offset of thread's float4
    const int bKc = (tid & 3) * 8;        // B stage: K offset of thread's chunk
    const int bNr = tid >> 2;             // B stage: n row 0..63 (+64 for i=1)

    for (int p = 0; p < 3; p++) {
        const float* Ap = g.A[p];
        if (!Ap) break;
        const int K = g.K[p];

        // per-pair staging plans (loop-invariant over k)
        const float* aPtr[4];
        bool aVal[4];
        int  aRow[4];
#pragma unroll
        for (int i = 0; i < 4; i++) {
            int r  = (tid >> 3) + i * 32;
            int gm = blockM + r;
            int grow = gm;
            if (g.rowMod[p] > 0) {        // rot < rowMod always
                grow = gm + g.rot[p];
                if (grow >= g.rowMod[p]) grow -= g.rowMod[p];
            }
            aRow[i] = r;
            aVal[i] = GUARD ? (gm < g.M) : true;
            aPtr[i] = Ap + (size_t)grow * K + aC4;
        }
        const __bf16* bPtr[2];
        int bRow[2];
#pragma unroll
        for (int i = 0; i < 2; i++) {
            int n   = bNr + i * 64;
            bRow[i] = n;
            bPtr[i] = g.B[p] + (size_t)(blockN + n) * K + bKc;
        }

        // ---- prologue: stage tile 0 into buffer 0 ----
        {
            float4 av[4];
            uint4  bv[2];
#pragma unroll
            for (int i = 0; i < 4; i++) {
                av[i] = make_float4(0.f, 0.f, 0.f, 0.f);
                if (!GUARD || aVal[i]) av[i] = *(const float4*)aPtr[i];
            }
#pragma unroll
            for (int i = 0; i < 2; i++) bv[i] = *(const uint4*)bPtr[i];
            __syncthreads();              // previous pair / launch reads done
#pragma unroll
            for (int i = 0; i < 4; i++) {
                v4bf pk = { (__bf16)av[i].x, (__bf16)av[i].y,
                            (__bf16)av[i].z, (__bf16)av[i].w };
                *(v4bf*)&As[0][aRow[i]][aC4] = pk;
            }
#pragma unroll
            for (int i = 0; i < 2; i++) *(uint4*)&Bt[0][bRow[i]][bKc] = bv[i];
        }

        // ---- main loop: one barrier per k-step (ping-pong) ----
        int cur = 0;
        for (int k0 = 0; k0 < K; k0 += BK) {
            __syncthreads();              // buffer[cur] stores now visible

            const bool more = (k0 + BK) < K;
            float4 av[4];
            if (more) {                   // A: issue next tile's loads early
#pragma unroll
                for (int i = 0; i < 4; i++) {
                    av[i] = make_float4(0.f, 0.f, 0.f, 0.f);
                    if (!GUARD || aVal[i]) {
                        av[i] = *(const float4*)(aPtr[i] + k0 + BK);
                        __builtin_prefetch(aPtr[i] + k0 + 2 * BK, 0, 1);
                    }
                }
            }

            // fragments: contiguous 2x b128 LDS reads each
            v16bf aF[2], bF[4];
#pragma unroll
            for (int mi = 0; mi < 2; mi++) {
                int rb = waveM * 32 + mi * 16 + l16;
                Frag u;
                u.q[0] = *(const uint4*)&As[cur][rb][half * 8];
                u.q[1] = *(const uint4*)&As[cur][rb][16 + half * 8];
                aF[mi] = u.v;
            }
#pragma unroll
            for (int ni = 0; ni < 4; ni++) {
                int cb = waveN * 64 + ni * 16 + l16;
                Frag u;
                u.q[0] = *(const uint4*)&Bt[cur][cb][half * 8];
                u.q[1] = *(const uint4*)&Bt[cur][cb][16 + half * 8];
                bF[ni] = u.v;
            }

#pragma unroll
            for (int mi = 0; mi < 2; mi++)
#pragma unroll
                for (int ni = 0; ni < 4; ni++)
                    acc[mi][ni] = __builtin_amdgcn_wmma_f32_16x16x32_bf16(
                        false, aF[mi], false, bF[ni], (short)0, acc[mi][ni],
                        false, false);

            if (more) {                   // store next tile into other buffer
#pragma unroll
                for (int i = 0; i < 4; i++) {
                    v4bf pk = { (__bf16)av[i].x, (__bf16)av[i].y,
                                (__bf16)av[i].z, (__bf16)av[i].w };
                    *(v4bf*)&As[cur ^ 1][aRow[i]][aC4] = pk;
                }
                // B: load late (L2-hot weights), store immediately
#pragma unroll
                for (int i = 0; i < 2; i++) {
                    uint4 bq = *(const uint4*)(bPtr[i] + k0 + BK);
                    *(uint4*)&Bt[cur ^ 1][bRow[i]][bKc] = bq;
                }
            }
            cur ^= 1;
        }
    }

    // epilogue: C/D layout — VGPR r holds M = r + 8*half, N = lane%16
#pragma unroll
    for (int ni = 0; ni < 4; ni++) {
        int col  = blockN + waveN * 64 + ni * 16 + l16;
        float bv = g.bias ? g.bias[col] : 0.0f;
#pragma unroll
        for (int mi = 0; mi < 2; mi++) {
            int rowBase = blockM + waveM * 32 + mi * 16 + half * 8;
            float* cPtr = g.C + (size_t)rowBase * g.N + col;
#pragma unroll
            for (int r = 0; r < 8; r++) {
                if (!GUARD || (rowBase + r < g.M)) {
                    float v = acc[mi][ni][r] + bv;
                    if (g.doRelu) v = fmaxf(v, 0.0f);
                    cPtr[(size_t)r * g.N] = v;
                }
            }
        }
    }
}

__global__ __launch_bounds__(256, 1)
void k_gemm_wmma(GemmArgs g) {
    __shared__ __bf16 As[2][BM][LDP];
    __shared__ __bf16 Bt[2][BN][LDP];
    if (blockIdx.y * BM + BM <= (unsigned)g.M)
        gemm_pairs<false>(g, As, Bt);     // interior tiles: no predication
    else
        gemm_pairs<true>(g, As, Bt);      // tail tile: row guards
}

// ---------------------------------------------------------------------------
// host orchestration
// ---------------------------------------------------------------------------

extern "C" void kernel_launch(void* const* d_in, const int* in_sizes, int n_in,
                              void* d_out, int out_size, void* d_ws, size_t ws_size,
                              hipStream_t stream) {
    (void)in_sizes; (void)n_in; (void)out_size; (void)ws_size;

    const float* L_init = (const float*)d_in[0];
    const float* C_init = (const float*)d_in[1];
    const float* lc_W   = (const float*)d_in[2];
    const float* lc_b   = (const float*)d_in[3];
    const float* cl_W   = (const float*)d_in[4];
    const float* cl_b   = (const float*)d_in[5];
    const float* lWx    = (const float*)d_in[6];
    const float* lWh    = (const float*)d_in[7];
    const float* lb     = (const float*)d_in[8];
    const float* cWx    = (const float*)d_in[9];
    const float* cWh    = (const float*)d_in[10];
    const float* cb     = (const float*)d_in[11];
    const float* v_Wh   = (const float*)d_in[12];
    const float* v_bh   = (const float*)d_in[13];
    const float* v_Wo   = (const float*)d_in[14];
    const float* v_bo   = (const float*)d_in[15];
    const int* edge_lits    = (const int*)d_in[16];
    const int* edge_clauses = (const int*)d_in[17];
    float* out = (float*)d_out;

    // ---- workspace layout ----
    char*  ws  = (char*)d_ws;
    size_t off = 0;
    auto abf = [&](size_t n) { __bf16* p = (__bf16*)(ws + off);
                               off = (off + n * 2 + 255) & ~(size_t)255; return p; };
    auto af32 = [&](size_t n) { float* p = (float*)(ws + off);
                                off = (off + n * 4 + 255) & ~(size_t)255; return p; };

    // transposed bf16 weights ([N][K] layout per slice)
    __bf16* lcWt = abf(3 * FM * FM);
    __bf16* clWt = abf(3 * FM * FM);
    __bf16* lWxt = abf(512 * 1024);     // two [1024][256] slices
    __bf16* lWht = abf(256 * 1024);
    __bf16* cWxt = abf(256 * 1024);
    __bf16* cWht = abf(256 * 1024);
    __bf16* vWt  = abf(2 * 512 * 512);  // [512][256]x2 then [512][512]

    float* lh  = af32((size_t)NL * FM);
    float* lcc = af32((size_t)NL * FM);
    float* ch  = af32((size_t)NC * FM);
    float* ccc = af32((size_t)NC * FM);
    float* t0  = af32((size_t)NC * FM);
    float* t1  = af32((size_t)NC * FM);
    float* t2  = af32((size_t)NC * FM);
    float* lcm = af32((size_t)NC * FM);
    float* clm = af32((size_t)NL * FM);
    float* Z   = af32((size_t)NC * 1024);   // shared for clause & literal gates

    auto convT = [&](const float* s, __bf16* d, int K, int N) {
        long n = (long)K * N;
        k_convT<<<(unsigned)((n + 255) / 256), 256, 0, stream>>>(s, d, K, N);
    };
    auto zero = [&](float* p, long n) {
        k_zero_f32<<<(unsigned)((n + 255) / 256), 256, 0, stream>>>(p, n);
    };
    // up to 3 fused (A,B) pairs accumulated into one output
    auto gemm3 = [&](const float* A0, int K0, int rot0, int mod0, const __bf16* B0,
                     const float* A1, int K1, int rot1, int mod1, const __bf16* B1,
                     const float* A2, int K2, int rot2, int mod2, const __bf16* B2,
                     int M, int N, const float* bias, float* C, int relu) {
        GemmArgs g;
        g.A[0] = A0; g.B[0] = B0; g.K[0] = K0; g.rot[0] = rot0; g.rowMod[0] = mod0;
        g.A[1] = A1; g.B[1] = B1; g.K[1] = K1; g.rot[1] = rot1; g.rowMod[1] = mod1;
        g.A[2] = A2; g.B[2] = B2; g.K[2] = K2; g.rot[2] = rot2; g.rowMod[2] = mod2;
        g.M = M; g.N = N; g.bias = bias; g.C = C; g.doRelu = relu;
        dim3 grid(N / BN, (M + BM - 1) / BM);
        k_gemm_wmma<<<grid, 256, 0, stream>>>(g);
    };
    auto gemm1 = [&](const float* A, int M, int K, const __bf16* B, int N,
                     const float* bias, float* C, int relu) {
        gemm3(A, K, 0, 0, B, nullptr, 0, 0, 0, nullptr, nullptr, 0, 0, 0, nullptr,
              M, N, bias, C, relu);
    };

    // ---- weight prep: f32 [K,N] -> bf16 [N,K], per GEMM slice ----
    for (int i = 0; i < 3; i++) {
        convT(lc_W + (size_t)i * FM * FM, lcWt + (size_t)i * FM * FM, FM, FM);
        convT(cl_W + (size_t)i * FM * FM, clWt + (size_t)i * FM * FM, FM, FM);
    }
    convT(lWx,              lWxt,              256, 1024);  // input-msg rows
    convT(lWx + 256 * 1024, lWxt + 256 * 1024, 256, 1024);  // flipped rows
    convT(lWh, lWht, 256, 1024);
    convT(cWx, cWxt, 256, 1024);
    convT(cWh, cWht, 256, 1024);
    convT(v_Wh,             vWt,             256, 512);     // layer0 top K-half
    convT(v_Wh + 256 * 512, vWt + 256 * 512, 256, 512);     // layer0 bottom K-half
    convT(v_Wh + 512 * 512, vWt + 512 * 512, 512, 512);     // layer1
    const __bf16* vW0a = vWt;
    const __bf16* vW0b = vWt + 256 * 512;
    const __bf16* vW1  = vWt + 512 * 512;

    // ---- state init: h = init / sqrt(FM), cells = 0 ----
    const float inv = 1.0f / 16.0f;  // 1/sqrt(256)
    k_bcast_row<<<(unsigned)(((long)NL * FM + 255) / 256), 256, 0, stream>>>(L_init, inv, lh, NL);
    k_bcast_row<<<(unsigned)(((long)NC * FM + 255) / 256), 256, 0, stream>>>(C_init, inv, ch, NC);
    zero(lcc, (long)NL * FM);
    zero(ccc, (long)NC * FM);

    // ---- message-passing rounds ----
    for (int r = 0; r < ROUNDS; r++) {
        // literal -> clause MLP (3 layers, relu on first two)
        gemm1(lh, NL, FM, lcWt,               FM, lc_b,          t0, 1);
        gemm1(t0, NL, FM, lcWt + FM * FM,     FM, lc_b + FM,     t1, 1);
        gemm1(t1, NL, FM, lcWt + 2 * FM * FM, FM, lc_b + 2 * FM, t2, 0);

        // segment-sum over edges into clauses
        zero(lcm, (long)NC * FM);
        k_scatter_add<<<NE, 256, 0, stream>>>(t2, edge_lits, edge_clauses, lcm, NE);

        // clause LSTM gates in one fused GEMM: Z = msgs@cWx + ch@cWh + cb
        gemm3(lcm, FM, 0, 0, cWxt,
              ch,  FM, 0, 0, cWht,
              nullptr, 0, 0, 0, nullptr,
              NC, 1024, cb, Z, 0);
        k_lstm<<<(unsigned)(((long)NC * FM + 255) / 256), 256, 0, stream>>>(Z, ch, ccc, NC);

        // clause -> literal MLP
        gemm1(ch, NC, FM, clWt,               FM, cl_b,          t0, 1);
        gemm1(t0, NC, FM, clWt + FM * FM,     FM, cl_b + FM,     t1, 1);
        gemm1(t1, NC, FM, clWt + 2 * FM * FM, FM, cl_b + 2 * FM, t2, 0);

        // segment-sum over edges into literals
        zero(clm, (long)NL * FM);
        k_scatter_add<<<NE, 256, 0, stream>>>(t2, edge_clauses, edge_lits, clm, NE);

        // literal LSTM gates in one fused GEMM:
        // Z = msgs@lWx[:256] + flipped@lWx[256:] + lh@lWh + lb
        gemm3(clm, FM, 0,  0,  lWxt,
              lh,  FM, NV, NL, lWxt + 256 * 1024,   // rotated rows = flipped
              lh,  FM, 0,  0,  lWht,
              NL, 1024, lb, Z, 0);
        k_lstm<<<(unsigned)(((long)NL * FM + 255) / 256), 256, 0, stream>>>(Z, lh, lcc, NL);
    }

    // ---- vote head: x = [lh_pos | lh_neg] -> relu MLP x2 -> linear ----
    // layer0 split along K: x@W0 = lh[:NV]@W0[0:256] + lh[NV:]@W0[256:512]
    gemm3(lh,                   FM, 0, 0, vW0a,
          lh + (size_t)NV * FM, FM, 0, 0, vW0b,
          nullptr, 0, 0, 0, nullptr,
          NV, 512, v_bh, t0, 1);
    gemm1(t0, NV, 512, vW1, 512, v_bh + 512, t1, 1);
    k_vote_out<<<(NV * 32 + 255) / 256, 256, 0, stream>>>(t1, v_Wo, v_bo, out, NV, 512);
}
